// FusedGromovWasserstein_14972255994598
// MI455X (gfx1250) — matrix operand order repaired and checked by reading, not compiled, not run on hardware
//
#include <hip/hip_runtime.h>
#include <hip/hip_bf16.h>

typedef __bf16 bf16_t;
typedef __attribute__((ext_vector_type(16))) __bf16 v16bf;
typedef __attribute__((ext_vector_type(8)))  float  v8f;
typedef __attribute__((ext_vector_type(2)))  float  v2f;
typedef __attribute__((ext_vector_type(4)))  unsigned u32x4;
typedef __attribute__((ext_vector_type(8)))  unsigned u32x8;

#define NB       8192
#define DDIM     256
#define LAMBDA   0.5f
#define TAUKL    0.1f
#define NOUTER   5
#define NSINK    10

union PackBF { unsigned u; bf16_t h[2]; };

// ---- prep: convert W1,W2 (f32 [256,256] row-major) to bf16 in workspace ----
__global__ void w_to_bf16_kernel(const float* __restrict__ W1,
                                 const float* __restrict__ W2,
                                 bf16_t* __restrict__ wbf) {
  int i = blockIdx.x * blockDim.x + threadIdx.x;
  if (i < 65536)       wbf[i] = (bf16_t)W1[i];
  else if (i < 131072) wbf[i] = (bf16_t)W2[i - 65536];
}

// ---- CDNA5 async/TDM helpers ----------------------------------------------
// One-shot TDM: 1-D tile of `nelem` f32 from global -> LDS (D# groups 0/1, 2-group form).
__device__ __forceinline__ void tdm_load_f32(unsigned lds_addr, const float* gaddr,
                                             unsigned nelem) {
  unsigned long long ga = (unsigned long long)(uintptr_t)gaddr;
  u32x4 g0;
  g0.x = __builtin_amdgcn_readfirstlane(1u);                       // count=1, user mode
  g0.y = __builtin_amdgcn_readfirstlane(lds_addr);                 // lds_addr
  g0.z = __builtin_amdgcn_readfirstlane((unsigned)ga);             // global_addr[31:0]
  g0.w = __builtin_amdgcn_readfirstlane(
           (((unsigned)(ga >> 32)) & 0x01FFFFFFu) | 0x80000000u);  // addr[56:32] | type=2
  u32x8 g1;
  g1.s0 = __builtin_amdgcn_readfirstlane(2u << 16);                // data_size=4B, wg_mask=0
  g1.s1 = __builtin_amdgcn_readfirstlane(nelem << 16);             // tensor_dim0[15:0]
  g1.s2 = __builtin_amdgcn_readfirstlane((nelem >> 16) | (1u << 16)); // dim0 hi | tensor_dim1=1
  g1.s3 = __builtin_amdgcn_readfirstlane(nelem << 16);             // tile_dim0 = nelem
  g1.s4 = __builtin_amdgcn_readfirstlane(1u);                      // tile_dim1 = 1
  g1.s5 = __builtin_amdgcn_readfirstlane(nelem);                   // tensor_dim0_stride lo
  g1.s6 = __builtin_amdgcn_readfirstlane(0u);
  g1.s7 = __builtin_amdgcn_readfirstlane(0u);
  asm volatile("tensor_load_to_lds %0, %1" :: "s"(g0), "s"(g1) : "memory");
}

__device__ __forceinline__ void async_copy16(unsigned lds_addr, const void* gaddr) {
  asm volatile("global_load_async_to_lds_b128 %0, %1, off"
               :: "v"(lds_addr), "v"(gaddr) : "memory");
}

// A fragment: 16x32 bf16, M=lane&15; VGPR p<4 -> K=2p(+8 hi lanes), p>=4 -> K=16+2(p-4)(+8)
__device__ __forceinline__ v16bf load_a_frag(const bf16_t* X, int lane, int ktile) {
  v16bf a;
  const int m   = lane & 15;
  const int hi8 = (lane >> 4) << 3;
  #pragma unroll
  for (int p = 0; p < 8; ++p) {
    const int kb = (p < 4) ? (p << 1) : (16 + ((p - 4) << 1));
    PackBF c;
    c.u = *(const unsigned*)(X + m * DDIM + ktile * 32 + kb + hi8);
    a[2 * p] = c.h[0];
    a[2 * p + 1] = c.h[1];
  }
  return a;
}

// B fragment: 32x16 bf16, B[k][n] = Wr[n][k]; N=lane&15, p<4 -> K=2p(+8 hi lanes), p>=4 -> K=16+2(p-4)(+8)
__device__ __forceinline__ v16bf load_b_frag(const bf16_t* Wr, int lane, int ntile, int ktile) {
  v16bf bf;
  const int n    = ntile * 16 + (lane & 15);
  const int hi8  = (lane >> 4) << 3;
  const bf16_t* row = Wr + n * DDIM + ktile * 32 + hi8;
  #pragma unroll
  for (int p = 0; p < 8; ++p) {
    PackBF c;
    c.u = *(const unsigned*)(row + ((p < 4) ? (p << 1) : (16 + ((p - 4) << 1))));
    bf[2 * p] = c.h[0];
    bf[2 * p + 1] = c.h[1];
  }
  return bf;
}

// Fused MLP layer for BOTH tensors: each W B-fragment is loaded once, used by 2 WMMAs.
__device__ __forceinline__ void gemm_layer2(const bf16_t* Xq, const bf16_t* Xr,
                                            const bf16_t* W, const float* __restrict__ bias,
                                            bf16_t* Yq, bf16_t* Yr,
                                            int wv, int lane, bool relu) {
  const int n    = lane & 15;
  const int moff = (lane >> 4) << 3;
  #pragma unroll
  for (int t = 0; t < 4; ++t) {
    const int ntile = wv * 4 + t;
    __builtin_prefetch(W + (((ntile + 1) & 15) * 16) * DDIM, 0, 1);
    v8f accq = {}, accr = {};
    #pragma unroll
    for (int kt = 0; kt < 8; ++kt) {
      v16bf bb = load_b_frag(W, lane, ntile, kt);
      v16bf a0 = load_a_frag(Xq, lane, kt);
      v16bf a1 = load_a_frag(Xr, lane, kt);
      accq = __builtin_amdgcn_wmma_f32_16x16x32_bf16(false, a0, false, bb, (short)0, accq, false, false);
      accr = __builtin_amdgcn_wmma_f32_16x16x32_bf16(false, a1, false, bb, (short)0, accr, false, false);
    }
    const float bv = bias[ntile * 16 + n];
    #pragma unroll
    for (int v = 0; v < 8; ++v) {
      float yq = accq[v] + bv;
      float yr = accr[v] + bv;
      if (relu) { yq = fmaxf(yq, 0.f); yr = fmaxf(yr, 0.f); }
      Yq[(v + moff) * DDIM + ntile * 16 + n] = (bf16_t)yq;
      Yr[(v + moff) * DDIM + ntile * 16 + n] = (bf16_t)yr;
    }
  }
}

// 16x16x16 f32 matmul via 4 chained V_WMMA_F32_16X16X4_F32, A/B read from LDS (f32 row-major)
__device__ __forceinline__ v8f mm16(const float* A, const float* Bm, int lane) {
  v8f acc = {};
  const int m   = lane & 15;
  const int hi2 = (lane >> 4) << 1;
  #pragma unroll
  for (int kt = 0; kt < 4; ++kt) {
    v2f a, bv;
    a[0]  = A[m * 16 + kt * 4 + hi2];
    a[1]  = A[m * 16 + kt * 4 + hi2 + 1];
    bv[0] = Bm[(kt * 4 + hi2) * 16 + m];
    bv[1] = Bm[(kt * 4 + hi2 + 1) * 16 + m];
    acc = __builtin_amdgcn_wmma_f32_16x16x4_f32(false, a, false, bv, (short)0, acc, false, false);
  }
  return acc;
}

#define LFENCE() asm volatile("" ::: "memory")

__global__ __launch_bounds__(128)
void fgw_main_kernel(const float* __restrict__ slots_q, const float* __restrict__ slots_r,
                     const float* __restrict__ mask_q,  const float* __restrict__ mask_r,
                     const float* __restrict__ cen_q,   const float* __restrict__ cen_r,
                     const float* __restrict__ b1,      const float* __restrict__ b2,
                     const float* __restrict__ log_eps,
                     const bf16_t* __restrict__ wbf,
                     float* __restrict__ out) {
  __shared__ bf16_t sb0[16 * DDIM], sb1[16 * DDIM], sb2[16 * DDIM], sb3[16 * DDIM];
  __shared__ float  scr[2048];
  __shared__ float  sf32[16 * DDIM];       // f32 staging tile for async/TDM copies

  const int b    = blockIdx.x;
  const int tid  = threadIdx.x;
  const int wv   = tid >> 5;
  const int lane = tid & 31;

  // ---- stage slots_q via Tensor Data Mover (one descriptor moves the 16KB tile) ----
  if (wv == 0) {
    tdm_load_f32((unsigned)(uintptr_t)sf32, slots_q + (size_t)b * 4096, 4096u);
    __builtin_amdgcn_s_wait_tensorcnt(0);
  }
  __syncthreads();
  #pragma unroll
  for (int i = 0; i < 8; ++i) {            // convert f32 -> bf16 (all 128 threads)
    int e = tid + i * 128;
    float4 v = ((const float4*)sf32)[e];
    sb0[e * 4 + 0] = (bf16_t)v.x; sb0[e * 4 + 1] = (bf16_t)v.y;
    sb0[e * 4 + 2] = (bf16_t)v.z; sb0[e * 4 + 3] = (bf16_t)v.w;
  }
  __syncthreads();

  // ---- stage slots_r via async global->LDS copies ----
  {
    const char* gr = (const char*)(slots_r + (size_t)b * 4096);
    const unsigned lbase = (unsigned)(uintptr_t)sf32;
    #pragma unroll
    for (int i = 0; i < 8; ++i) {
      const unsigned off = (unsigned)(tid + i * 128) * 16u;
      async_copy16(lbase + off, gr + off);
    }
    asm volatile("s_wait_asynccnt 0x0" ::: "memory");
  }
  __syncthreads();
  #pragma unroll
  for (int i = 0; i < 8; ++i) {
    int e = tid + i * 128;
    float4 v = ((const float4*)sf32)[e];
    sb1[e * 4 + 0] = (bf16_t)v.x; sb1[e * 4 + 1] = (bf16_t)v.y;
    sb1[e * 4 + 2] = (bf16_t)v.z; sb1[e * 4 + 3] = (bf16_t)v.w;
  }
  __syncthreads();

  // ---- projection MLP: 2 layers, q and r fused so each W fragment feeds 2 WMMAs ----
  gemm_layer2(sb0, sb1, wbf,         b1, sb2, sb3, wv, lane, true);
  __syncthreads();
  gemm_layer2(sb2, sb3, wbf + 65536, b2, sb0, sb1, wv, lane, false);   // pq->sb0, pr->sb1
  __syncthreads();

  if (wv != 0) return;   // FGW loop is a single-wave 16x16 problem

  float* sT   = scr;
  float* sSq  = scr + 256;
  float* sSr  = scr + 512;
  float* sSq2 = scr + 768;
  float* sSr2 = scr + 1024;
  float* sP   = scr + 1280;
  float* sLK  = scr + 1536;
  float* srs  = scr + 1792;
  float* scs  = scr + 1808;
  float* st1  = scr + 1824;
  float* st2  = scr + 1840;
  float* slu  = scr + 1856;
  float* slv  = scr + 1872;
  float* slmu = scr + 1888;
  float* slnu = scr + 1904;
  float* smu  = scr + 1920;
  float* snuv = scr + 1936;
  float* snq  = scr + 1952;
  float* snr  = scr + 1968;
  float* sred = scr + 1984;

  const int n    = lane & 15;
  const int moff = (lane >> 4) << 3;

  // ---- norms of projections ----
  if (lane < 16) {
    float s = 0.f;
    for (int d = 0; d < DDIM; ++d) { float x = (float)sb0[lane * DDIM + d]; s += x * x; }
    snq[lane] = s;
  } else {
    const int l = lane - 16;
    float s = 0.f;
    for (int d = 0; d < DDIM; ++d) { float x = (float)sb1[l * DDIM + d]; s += x * x; }
    snr[l] = s;
  }
  LFENCE();

  // ---- C = pdist(pq, pr): Gram via bf16 WMMA ----
  v8f g = {};
  #pragma unroll
  for (int kt = 0; kt < 8; ++kt) {
    v16bf a  = load_a_frag(sb0, lane, kt);
    v16bf bb = load_b_frag(sb1, lane, 0, kt);   // B[k][n] = pr[n][k]
    g = __builtin_amdgcn_wmma_f32_16x16x32_bf16(false, a, false, bb, (short)0, g, false, false);
  }
  v8f Cd;
  {
    const float nrn = snr[n];
    #pragma unroll
    for (int v = 0; v < 8; ++v) {
      float d2 = snq[v + moff] + nrn - 2.f * g[v];
      Cd[v] = sqrtf(fmaxf(d2, 1e-6f));
    }
  }

  // ---- Sq, Sr from 2-D centroids (D-layout, stored to LDS) ----
  {
    const float* cq = cen_q + (size_t)b * 32;
    const float* cr = cen_r + (size_t)b * 32;
    const float qnx = cq[2 * n], qny = cq[2 * n + 1];
    const float rnx = cr[2 * n], rny = cr[2 * n + 1];
    #pragma unroll
    for (int v = 0; v < 8; ++v) {
      const int m = v + moff;
      float dx = cq[2 * m] - qnx, dy = cq[2 * m + 1] - qny;
      float d2 = fmaxf(dx * dx + dy * dy, 1e-6f);
      sSq2[m * 16 + n] = d2;  sSq[m * 16 + n] = sqrtf(d2);
      dx = cr[2 * m] - rnx;  dy = cr[2 * m + 1] - rny;
      d2 = fmaxf(dx * dx + dy * dy, 1e-6f);
      sSr2[m * 16 + n] = d2;  sSr[m * 16 + n] = sqrtf(d2);
    }
  }

  // ---- mu, nu from masks ----
  if (lane < 16) { smu[lane] = mask_q[b * 16 + lane]; snuv[lane] = mask_r[b * 16 + lane]; }
  LFENCE();
  {
    float sq_ = 0.f, sr_ = 0.f;
    for (int j = 0; j < 16; ++j) { sq_ += smu[j]; sr_ += snuv[j]; }
    if (lane < 16) {
      float mu = smu[lane] / (sq_ + 1e-8f);
      float nu = snuv[lane] / (sr_ + 1e-8f);
      smu[lane] = mu;  snuv[lane] = nu;
      slmu[lane] = logf(fmaxf(mu, 1e-8f));
      slnu[lane] = logf(fmaxf(nu, 1e-8f));
    }
  }
  LFENCE();

  float eps = expf(log_eps[0]);
  eps = fminf(fmaxf(eps, 0.01f), 0.5f);
  const float rho = TAUKL / (TAUKL + eps);

  v8f Td;
  {
    const float nun = snuv[n];
    #pragma unroll
    for (int v = 0; v < 8; ++v) Td[v] = smu[v + moff] * nun;
  }

  v8f Cfd = {}, LKd = {};
  for (int it = 0; it < NOUTER; ++it) {
    #pragma unroll
    for (int v = 0; v < 8; ++v) sT[(v + moff) * 16 + n] = Td[v];
    LFENCE();
    if (lane < 16) {                       // row sums of T
      float s = 0.f; for (int j = 0; j < 16; ++j) s += sT[lane * 16 + j];
      srs[lane] = s;
    } else {                               // col sums of T
      const int c = lane - 16;
      float s = 0.f; for (int j = 0; j < 16; ++j) s += sT[j * 16 + c];
      scs[c] = s;
    }
    LFENCE();
    if (lane < 16) {                       // t1 = Sq2 @ rowsum
      float s = 0.f; for (int j = 0; j < 16; ++j) s += sSq2[lane * 16 + j] * srs[j];
      st1[lane] = s;
    } else {                               // t2 = colsum @ Sr2
      const int c = lane - 16;
      float s = 0.f; for (int j = 0; j < 16; ++j) s += scs[j] * sSr2[j * 16 + c];
      st2[c] = s;
    }
    LFENCE();

    v8f P = mm16(sSq, sT, lane);           // Sq @ T   (f32 WMMA)
    #pragma unroll
    for (int v = 0; v < 8; ++v) sP[(v + moff) * 16 + n] = P[v];
    LFENCE();
    v8f G3 = mm16(sP, sSr, lane);          // (Sq@T) @ Sr

    const float t2n = st2[n];
    #pragma unroll
    for (int v = 0; v < 8; ++v) {
      const float lgw = st1[v + moff] + t2n - 2.f * G3[v];
      Cfd[v] = (1.f - LAMBDA) * Cd[v] + LAMBDA * lgw;
      LKd[v] = -Cfd[v] / eps;
      sLK[(v + moff) * 16 + n] = LKd[v];
    }
    if (lane < 16) { slu[lane] = 0.f; slv[lane] = 0.f; }
    LFENCE();

    for (int s_ = 0; s_ < NSINK; ++s_) {
      if (lane < 16) {
        const int m = lane;
        float mx = -1e30f;
        #pragma unroll
        for (int j = 0; j < 16; ++j) mx = fmaxf(mx, sLK[m * 16 + j] + slv[j]);
        float s = 0.f;
        #pragma unroll
        for (int j = 0; j < 16; ++j) s += expf(sLK[m * 16 + j] + slv[j] - mx);
        slu[m] = rho * (slmu[m] - (mx + logf(s)));
      }
      LFENCE();
      if (lane < 16) {
        const int c = lane;
        float mx = -1e30f;
        #pragma unroll
        for (int j = 0; j < 16; ++j) mx = fmaxf(mx, sLK[j * 16 + c] + slu[j]);
        float s = 0.f;
        #pragma unroll
        for (int j = 0; j < 16; ++j) s += expf(sLK[j * 16 + c] + slu[j] - mx);
        slv[c] = rho * (slnu[c] - (mx + logf(s)));
      }
      LFENCE();
    }

    const float lvn = slv[n];
    #pragma unroll
    for (int v = 0; v < 8; ++v) Td[v] = expf(slu[v + moff] + LKd[v] + lvn);
  }

  // ---- cost, outputs ----
  float part = 0.f;
  #pragma unroll
  for (int v = 0; v < 8; ++v) part += Td[v] * Cfd[v];
  sred[lane] = part;
  LFENCE();

  float* outT = out + NB + (size_t)b * 256;
  float* outC = out + NB + (size_t)NB * 256 + (size_t)b * 256;
  #pragma unroll
  for (int v = 0; v < 8; ++v) {
    outT[(v + moff) * 16 + n] = Td[v];
    outC[(v + moff) * 16 + n] = Cd[v];
  }
  if (lane == 0) {
    float c = 0.f;
    for (int j = 0; j < 32; ++j) c += sred[j];
    out[NB + 2 * (size_t)NB * 256 + b] = c;       // cost
    out[b] = 1.f / (1.f + expf(c));               // sigmoid(-cost)
  }
}

extern "C" void kernel_launch(void* const* d_in, const int* in_sizes, int n_in,
                              void* d_out, int out_size, void* d_ws, size_t ws_size,
                              hipStream_t stream) {
  (void)in_sizes; (void)n_in; (void)out_size; (void)ws_size;
  const float* slots_q = (const float*)d_in[0];
  const float* slots_r = (const float*)d_in[1];
  const float* mask_q  = (const float*)d_in[2];
  const float* mask_r  = (const float*)d_in[3];
  const float* cen_q   = (const float*)d_in[4];
  const float* cen_r   = (const float*)d_in[5];
  const float* W1      = (const float*)d_in[6];
  const float* b1      = (const float*)d_in[7];
  const float* W2      = (const float*)d_in[8];
  const float* b2      = (const float*)d_in[9];
  const float* log_eps = (const float*)d_in[10];
  bf16_t* wbf = (bf16_t*)d_ws;   // needs 256 KB of workspace

  w_to_bf16_kernel<<<512, 256, 0, stream>>>(W1, W2, wbf);
  fgw_main_kernel<<<NB, 128, 0, stream>>>(slots_q, slots_r, mask_q, mask_r,
                                          cen_q, cen_r, b1, b2, log_eps, wbf,
                                          (float*)d_out);
}
